// atn_style_19980187861169
// MI455X (gfx1250) — matrix operand order repaired and checked by reading, not compiled, not run on hardware
//
#include <hip/hip_runtime.h>
#include <cstddef>
#include <cstdint>

// ---------------------------------------------------------------------------
// Problem constants (from reference setup_inputs)
// ---------------------------------------------------------------------------
#define BB 32      // batch
#define CC 256     // channels (== Lx)
#define LX 256     // content length
#define LS 8192    // style length
#define EPS 1e-5f

// attention tiling
#define LT 64              // query (l) tile per workgroup  -> grid.y = 4
#define MT 64              // key   (m) tile per iteration
#define CCP 264            // stride for [*][c]-style rows (mult of 8 -> 16B align)
#define MTP 72             // stride for [c][m] / [l][m] rows (mult of 8)

typedef __bf16 bf16;
typedef __attribute__((ext_vector_type(16))) __bf16 v16bf;
typedef __attribute__((ext_vector_type(8)))  __bf16 v8bf;
typedef __attribute__((ext_vector_type(4)))  __bf16 v4bf;
typedef __attribute__((ext_vector_type(8)))  float  v8f;
typedef __attribute__((ext_vector_type(4)))  unsigned int u32x4;
typedef __attribute__((ext_vector_type(4)))  int i32x4;
typedef __attribute__((ext_vector_type(8)))  int i32x8;

// ---------------------------------------------------------------------------
// WMMA fragment helpers (gfx1250 wave32, CDNA5 ISA layouts).
// A 16x32 bf16 frag: lane row = lane&15; f[0..7]  = K kh*8+0..7,
//                                        f[8..15] = K 16+kh*8+0..7 (kh=lane>>4)
//   -> row-major storage gives two contiguous 16B loads per lane.
// B 32x16 bf16 frag: lane col = lane&15; f[0..7]  = K kh*8+0..7,
//                                        f[8..15] = K 16+kh*8+0..7
//   -> N-major storage ([col][k]) gives two contiguous 16B loads per lane.
// C/D f32: VGPR r -> row M = r + 8*(lane>=16), col N = lane&15.
// ---------------------------------------------------------------------------
__device__ inline v16bf frag_A_rm(const bf16* __restrict__ p, int stride) {
  const int lane = threadIdx.x & 31;
  const bf16* r = p + (lane & 15) * stride + (lane >> 4) * 8;
  v8bf lo = *(const v8bf*)r;        // K run kh*8 .. +7
  v8bf hi = *(const v8bf*)(r + 16); // K run 16+kh*8 .. +7
  return __builtin_shufflevector(lo, hi, 0, 1, 2, 3, 4, 5, 6, 7,
                                 8, 9, 10, 11, 12, 13, 14, 15);
}

__device__ inline v16bf frag_B_nm(const bf16* __restrict__ p, int stride) {
  const int lane = threadIdx.x & 31;
  const bf16* r = p + (lane & 15) * stride + (lane >> 4) * 8;
  v8bf lo = *(const v8bf*)r;        // K run kh*8 .. +7
  v8bf hi = *(const v8bf*)(r + 16); // K run 16+kh*8 .. +7
  return __builtin_shufflevector(lo, hi, 0, 1, 2, 3, 4, 5, 6, 7,
                                 8, 9, 10, 11, 12, 13, 14, 15);
}

__device__ inline v8f wmma_bf16(v16bf a, v16bf b, v8f c) {
  return __builtin_amdgcn_wmma_f32_16x16x32_bf16(false, a, false, b,
                                                 (short)0, c, false, false);
}

// ---------------------------------------------------------------------------
// Stage 1: instance-norm statistics
// ---------------------------------------------------------------------------
__global__ __launch_bounds__(256) void k_stats_g(const float* __restrict__ s_sty,
                                                 float* __restrict__ gsum,
                                                 float* __restrict__ gsumsq) {
  const int b = blockIdx.x;
  const int mc = blockIdx.y;
  const int c = threadIdx.x;
  float s = 0.f, s2 = 0.f;
  for (int m = mc * 256; m < mc * 256 + 256; ++m) {
    float v = s_sty[((size_t)m * BB + b) * CC + c];
    s += v; s2 += v * v;
  }
  atomicAdd(&gsum[b * CC + c], s);
  atomicAdd(&gsumsq[b * CC + c], s2);
}

__global__ __launch_bounds__(256) void k_finalize_g(const float* __restrict__ gsum,
                                                    const float* __restrict__ gsumsq,
                                                    float* __restrict__ muG,
                                                    float* __restrict__ rstdG) {
  const int i = blockIdx.x * CC + threadIdx.x;
  float mu = gsum[i] * (1.f / LS);
  float var = gsumsq[i] * (1.f / LS) - mu * mu;
  muG[i] = mu;
  rstdG[i] = rsqrtf(var + EPS);
}

__global__ __launch_bounds__(256) void k_stats_f(const float* __restrict__ x,
                                                 float* __restrict__ muF,
                                                 float* __restrict__ rstdF) {
  const int b = blockIdx.x;
  const int c = threadIdx.x;
  float s = 0.f, s2 = 0.f;
  for (int l = 0; l < LX; ++l) {
    float v = x[((size_t)l * BB + b) * CC + c];
    s += v; s2 += v * v;
  }
  float mu = s * (1.f / LX);
  float var = s2 * (1.f / LX) - mu * mu;
  muF[b * CC + c] = mu;
  rstdF[b * CC + c] = rsqrtf(var + EPS);
}

// ---------------------------------------------------------------------------
// Stage 2: weight preparation
// ---------------------------------------------------------------------------
__global__ __launch_bounds__(256) void k_wconv(const float* __restrict__ wf,
                                               const float* __restrict__ wh,
                                               const float* __restrict__ wk,
                                               float* __restrict__ wft,
                                               bf16* __restrict__ whb,
                                               bf16* __restrict__ wkb) {
  const int o = blockIdx.x;
  const int c = threadIdx.x;
  wft[c * CC + o] = wf[o * CC + c];
  whb[o * CC + c] = (bf16)wh[o * CC + c];
  wkb[o * CC + c] = (bf16)wk[o * CC + c];
}

// Fold instance-norm into conv-g:  Gp = (Wg*rstd) Gm + (bg - Wg*(mu*rstd))
__global__ __launch_bounds__(256) void k_prep_wg(const float* __restrict__ wg,
                                                 const float* __restrict__ bg,
                                                 const float* __restrict__ muG,
                                                 const float* __restrict__ rstdG,
                                                 bf16* __restrict__ wgeff,
                                                 float* __restrict__ bgeff) {
  const int b = blockIdx.x;
  const int o = blockIdx.y;
  const int c = threadIdx.x;
  float we = wg[o * CC + c] * rstdG[b * CC + c];
  wgeff[((size_t)(b * CC + o)) * CC + c] = (bf16)we;
  __shared__ float red[256];
  red[c] = we * muG[b * CC + c];
  __syncthreads();
  for (int s = 128; s; s >>= 1) {
    if (c < s) red[c] += red[c + s];
    __syncthreads();
  }
  if (c == 0) bgeff[b * CC + o] = bg[o] - red[0];
}

// Fp^T[b][l][o] = sum_c Wf[o,c]*(x[l,b,c]-mu)*rstd + bf[o]
__global__ __launch_bounds__(256) void k_fpt(const float* __restrict__ x,
                                             const float* __restrict__ wft,
                                             const float* __restrict__ bfv,
                                             const float* __restrict__ muF,
                                             const float* __restrict__ rstdF,
                                             bf16* __restrict__ fpt) {
  const int b = blockIdx.x / LX;
  const int l = blockIdx.x % LX;
  const int t = threadIdx.x;
  __shared__ float xn[CC];
  float xv = x[((size_t)l * BB + b) * CC + t];
  xn[t] = (xv - muF[b * CC + t]) * rstdF[b * CC + t];
  __syncthreads();
  float acc = bfv[t];
  for (int c = 0; c < CC; ++c) acc += wft[c * CC + t] * xn[c];
  fpt[((size_t)b * LX + l) * CC + t] = (bf16)acc;
}

// ---------------------------------------------------------------------------
// Stage 3: fused conv-g/conv-h + flash attention over style positions.
// s_sty tile is fetched by the Tensor Data Mover (TDM) straight into LDS:
// 2D descriptor, rows (mb+y)*B + b via tensor_dim0_stride = B*C, with the
// D# pad fields generating the 264-dword padded row stride in hardware.
// LDS layouts keep every WMMA fragment load as 2x contiguous 16B:
//   s_fpt [l][c]   row-major  -> GEMM2 A
//   s_gm  [m][c]   N-major    -> GEMM1 B
//   s_gp  [m][o]   N-major    -> GEMM2 B    (packed v8bf D-stores)
//   s_h   [c][m]   row-major  -> GEMM3 A    (strided b16 D-stores)
//   s_ptT [l][m]   N-major    -> GEMM3 B    (strided b16 P-stores)
// The f32 TDM staging tile aliases s_gp/s_h (dead across phases).
// ---------------------------------------------------------------------------
#define SMEM_FPT 0
#define SMEM_GM  33792
#define SMEM_GPH 67584          // staging f32 tile aliases s_gp + s_h
#define SMEM_SZ  (67584 + 33792 + 36864)

__global__ __launch_bounds__(256) void k_attn(const float* __restrict__ s_sty,
                                              const bf16* __restrict__ fpt,
                                              const bf16* __restrict__ wgeff,
                                              const float* __restrict__ bgeff,
                                              const bf16* __restrict__ whb,
                                              const float* __restrict__ bh,
                                              bf16* __restrict__ otb) {
  const int b    = blockIdx.x;
  const int l0   = blockIdx.y * LT;
  const int tid  = threadIdx.x;
  const int wave = tid >> 5;
  const int lane = tid & 31;
  const int col16 = lane & 15;
  const int kh    = lane >> 4;

  __shared__ __align__(16) char smem[SMEM_SZ];
  bf16*  s_fpt  = (bf16*)(smem + SMEM_FPT);          // [LT][CCP]
  bf16*  s_gm   = (bf16*)(smem + SMEM_GM);           // [MT][CCP]
  bf16*  s_gp   = (bf16*)(smem + SMEM_GPH);          // [MT][CCP]
  bf16*  s_h    = (bf16*)(smem + SMEM_GPH + 33792);  // [CC][MTP]
  float* stagef = (float*)(smem + SMEM_GPH);         // [MT][CCP] f32 (alias)
  bf16*  s_ptT  = s_gm;                              // [LT][MTP] (alias)
  __shared__ float sm_rowmax[LT];
  __shared__ float sm_rowsum[LT];
  __shared__ float sm_pmax[2][LT];
  __shared__ float sm_psum[2][LT];
  __shared__ float sm_scale[LT];

  for (int idx = tid; idx < LT * (CC / 4); idx += 256) {
    int l = idx >> 6, cq = idx & 63;
    *(v4bf*)&s_fpt[l * CCP + 4 * cq] =
        *(const v4bf*)&fpt[((size_t)b * LX + (l0 + l)) * CC + 4 * cq];
  }
  if (tid < LT) { sm_rowmax[tid] = -3.4e38f; sm_rowsum[tid] = 0.f; }

  v8f accO[2][4] = {};                // O: c rows [32w,32w+32) x l cols [0,64)
  const int c0 = wave * 32;
  const int lt = wave & 3;            // logit l-tile of this wave
  const int mh = wave >> 2;           // logit m-half of this wave

  for (int mb = 0; mb < LS; mb += MT) {
    // ---- TDM: DMA (MT x CC) f32 tile into LDS staging; wave 0 issues ----
    if (wave == 0) {
      unsigned long long ga =
          (unsigned long long)(uintptr_t)(s_sty + ((size_t)mb * BB + b) * CC);
      unsigned int lds_off = (unsigned int)(uintptr_t)stagef;  // LDS offset = addr[31:0]
      u32x4 g0;
      g0[0] = 1u;                                            // count=1 (valid user D#)
      g0[1] = lds_off;                                       // lds_addr
      g0[2] = (unsigned int)(ga & 0xffffffffu);              // global_addr[31:0]
      g0[3] = (unsigned int)((ga >> 32) & 0x1ffffffu) | (2u << 30); // [56:32] | type=2
      i32x8 g1;
      g1[0] = (int)((2u << 16) | (1u << 20) | (7u << 22) | (7u << 25));
      //          data_size=4B | pad_en | pad_interval=256dw | pad_amount=8dw
      g1[1] = (int)(256u << 16);           // tensor_dim0 = 256 (low 16 bits)
      g1[2] = 0;                           // dim0 hi / dim1 lo16
      g1[3] = (int)(4u | (256u << 16));    // tensor_dim1 hi16 (=262144) | tile_dim0=256
      g1[4] = (int)64u;                    // tile_dim1 = 64, tile_dim2 = 0
      g1[5] = (int)8192u;                  // tensor_dim0_stride = B*C elems
      g1[6] = 0;
      g1[7] = 0;
      i32x4 z4 = {0, 0, 0, 0};
#if defined(__clang_major__) && __clang_major__ >= 23
      i32x8 z8 = {0, 0, 0, 0, 0, 0, 0, 0};
      __builtin_amdgcn_tensor_load_to_lds(g0, g1, z4, z4, z8, 0);
#else
      __builtin_amdgcn_tensor_load_to_lds(g0, g1, z4, z4, 0);
#endif
      __builtin_amdgcn_s_wait_tensorcnt(0);
    }
    __syncthreads();

    // ---- convert staged f32 tile -> bf16 s_gm[m][c]; prefetch next tile ----
    for (int idx = tid; idx < MT * (CC / 4); idx += 256) {
      int m = idx >> 6, cq = idx & 63;
      float4 v = *(const float4*)&stagef[m * CCP + 4 * cq];
      v4bf o; o[0] = (bf16)v.x; o[1] = (bf16)v.y; o[2] = (bf16)v.z; o[3] = (bf16)v.w;
      *(v4bf*)&s_gm[m * CCP + 4 * cq] = o;
      if (mb + MT < LS)
        __builtin_prefetch(&s_sty[((size_t)(mb + MT + m) * BB + b) * CC + 4 * cq], 0, 1);
    }
    __syncthreads();

    // ---- GEMM1: Gp = WgEff(b)*Gm + bgEff ; H = Wh*Gm + bh ----
    {
      v8f aG[2][4] = {}, aH[2][4] = {};
      for (int kk = 0; kk < CC; kk += 32) {
        v16bf bfr[4];
#pragma unroll
        for (int j = 0; j < 4; ++j)
          bfr[j] = frag_B_nm(s_gm + (16 * j) * CCP + kk, CCP);
#pragma unroll
        for (int i = 0; i < 2; ++i) {
          v16bf ag = frag_A_rm(wgeff + ((size_t)(b * CC + c0 + 16 * i)) * CC + kk, CC);
          v16bf ah = frag_A_rm(whb + (size_t)(c0 + 16 * i) * CC + kk, CC);
#pragma unroll
          for (int j = 0; j < 4; ++j) {
            aG[i][j] = wmma_bf16(ag, bfr[j], aG[i][j]);
            aH[i][j] = wmma_bf16(ah, bfr[j], aH[i][j]);
          }
        }
      }
#pragma unroll
      for (int i = 0; i < 2; ++i)
#pragma unroll
        for (int j = 0; j < 4; ++j) {
          const int m = 16 * j + col16;        // lane's key column
          const int ob = c0 + 16 * i + 8 * kh; // 8-consecutive output channels
          v8bf gq;
#pragma unroll
          for (int r = 0; r < 8; ++r) gq[r] = (bf16)(aG[i][j][r] + bgeff[b * CC + ob + r]);
          *(v8bf*)&s_gp[m * CCP + ob] = gq;    // packed ds_store_b128
#pragma unroll
          for (int r = 0; r < 8; ++r)          // strided b16 (transpose to [c][m])
            s_h[(ob + r) * MTP + m] = (bf16)(aH[i][j][r] + bh[ob + r]);
        }
    }
    __syncthreads();

    // ---- GEMM2: logits = Fp^T * Gp  (wave: l-tile lt, m cols [32mh,32mh+32)) ----
    v8f S[2] = {};
    for (int kk = 0; kk < CC; kk += 32) {
      v16bf a = frag_A_rm(s_fpt + (16 * lt) * CCP + kk, CCP);
#pragma unroll
      for (int j = 0; j < 2; ++j) {
        v16bf bb = frag_B_nm(s_gp + (16 * (2 * mh + j)) * CCP + kk, CCP);
        S[j] = wmma_bf16(a, bb, S[j]);
      }
    }

    // ---- online softmax: cross-lane (width-16) + cross-wave via LDS ----
    float pm[8];
#pragma unroll
    for (int r = 0; r < 8; ++r) {
      pm[r] = fmaxf(S[0][r], S[1][r]);
#pragma unroll
      for (int off = 8; off; off >>= 1)
        pm[r] = fmaxf(pm[r], __shfl_xor(pm[r], off, 16));
    }
    if (col16 == 0)
#pragma unroll
      for (int r = 0; r < 8; ++r)
        sm_pmax[mh][16 * lt + r + 8 * kh] = pm[r];
    __syncthreads();
    if (tid < LT) {
      float mold = sm_rowmax[tid];
      float mnew = fmaxf(mold, fmaxf(sm_pmax[0][tid], sm_pmax[1][tid]));
      sm_scale[tid] = __expf(mold - mnew);
      sm_rowmax[tid] = mnew;
      sm_psum[0][tid] = 0.f;
      sm_psum[1][tid] = 0.f;
    }
    __syncthreads();

    float ps[8] = {0, 0, 0, 0, 0, 0, 0, 0};
#pragma unroll
    for (int j = 0; j < 2; ++j) {
      const int m = 32 * mh + 16 * j + col16;          // lane's key column
#pragma unroll
      for (int r = 0; r < 8; ++r) {
        int row = 16 * lt + r + 8 * kh;                // local l
        float p = __expf(S[j][r] - sm_rowmax[row]);
        ps[r] += p;
        s_ptT[row * MTP + m] = (bf16)p;                // strided b16 (-> [l][m])
      }
    }
#pragma unroll
    for (int r = 0; r < 8; ++r)
#pragma unroll
      for (int off = 8; off; off >>= 1)
        ps[r] += __shfl_xor(ps[r], off, 16);
    if (col16 == 0)
#pragma unroll
      for (int r = 0; r < 8; ++r)
        sm_psum[mh][16 * lt + r + 8 * kh] = ps[r];
    __syncthreads();
    if (tid < LT)
      sm_rowsum[tid] = sm_rowsum[tid] * sm_scale[tid] + sm_psum[0][tid] + sm_psum[1][tid];
    __syncthreads();

    // ---- rescale O, then GEMM3: O += H * P^T ----
#pragma unroll
    for (int i = 0; i < 2; ++i)
#pragma unroll
      for (int jt = 0; jt < 4; ++jt) {
        float sc = sm_scale[16 * jt + col16];
#pragma unroll
        for (int r = 0; r < 8; ++r) accO[i][jt][r] *= sc;
      }
    for (int kb = 0; kb < MT; kb += 32) {
      v16bf bfr[4];
#pragma unroll
      for (int jt = 0; jt < 4; ++jt)
        bfr[jt] = frag_B_nm(s_ptT + (16 * jt) * MTP + kb, MTP);
#pragma unroll
      for (int i = 0; i < 2; ++i) {
        v16bf a = frag_A_rm(s_h + (c0 + 16 * i) * MTP + kb, MTP);
#pragma unroll
        for (int jt = 0; jt < 4; ++jt)
          accO[i][jt] = wmma_bf16(a, bfr[jt], accO[i][jt]);
      }
    }
    __syncthreads();
  }

  // ---- normalize, store O^T as (b, l, c) bf16 (packed 16B stores) ----
#pragma unroll
  for (int i = 0; i < 2; ++i)
#pragma unroll
    for (int jt = 0; jt < 4; ++jt) {
      int lloc = 16 * jt + col16;
      float inv = 1.0f / sm_rowsum[lloc];
      v8bf o;
#pragma unroll
      for (int r = 0; r < 8; ++r) o[r] = (bf16)(accO[i][jt][r] * inv);
      *(v8bf*)&otb[((size_t)b * LX + (l0 + lloc)) * CC + c0 + 16 * i + 8 * kh] = o;
    }
}

// ---------------------------------------------------------------------------
// Stage 4: Ok = Wk * O^T per batch (256^3 GEMM), add bk, residual x.
// out[i, b, j] = sum_l Wk[j,l] * Ot[b,l,i] + bk[j] + x[i,b,j]
// LDS panel staged transposed ([i][l]) so B-fragments are contiguous.
// ---------------------------------------------------------------------------
#define LP2 40
__global__ __launch_bounds__(256) void k_final(const bf16* __restrict__ wkb,
                                               const bf16* __restrict__ otb,
                                               const float* __restrict__ bk,
                                               const float* __restrict__ x,
                                               float* __restrict__ out) {
  const int b  = blockIdx.x;
  const int jh = blockIdx.y;                  // 0..1 -> j-half
  const int tid = threadIdx.x;
  const int wave = tid >> 5, lane = tid & 31;
  const int col16 = lane & 15, kh = lane >> 4;
  const int j0 = jh * 128 + wave * 16;

  __shared__ bf16 s_ot[CC * LP2];             // O^T panel, transposed to [i][l]
  v8f acc[16] = {};
  for (int kk = 0; kk < CC; kk += 32) {
    for (int idx = tid; idx < 32 * CC; idx += 256) {
      int k = idx >> 8, i = idx & 255;
      s_ot[i * LP2 + k] = otb[((size_t)b * LX + kk + k) * CC + i];
    }
    __syncthreads();
    v16bf a = frag_A_rm(wkb + (size_t)j0 * CC + kk, CC);
#pragma unroll
    for (int t = 0; t < 16; ++t)
      acc[t] = wmma_bf16(a, frag_B_nm(s_ot + (16 * t) * LP2, LP2), acc[t]);
    __syncthreads();
  }
#pragma unroll
  for (int t = 0; t < 16; ++t) {
    int i = 16 * t + col16;
    size_t o0 = ((size_t)i * BB + b) * CC + j0 + 8 * kh;
    float4 lo = *(const float4*)&x[o0];
    float4 hi = *(const float4*)&x[o0 + 4];
    float4 r0, r1;
    r0.x = acc[t][0] + bk[j0 + 8 * kh + 0] + lo.x;
    r0.y = acc[t][1] + bk[j0 + 8 * kh + 1] + lo.y;
    r0.z = acc[t][2] + bk[j0 + 8 * kh + 2] + lo.z;
    r0.w = acc[t][3] + bk[j0 + 8 * kh + 3] + lo.w;
    r1.x = acc[t][4] + bk[j0 + 8 * kh + 4] + hi.x;
    r1.y = acc[t][5] + bk[j0 + 8 * kh + 5] + hi.y;
    r1.z = acc[t][6] + bk[j0 + 8 * kh + 6] + hi.z;
    r1.w = acc[t][7] + bk[j0 + 8 * kh + 7] + hi.w;
    *(float4*)&out[o0] = r0;
    *(float4*)&out[o0 + 4] = r1;
  }
}

// ---------------------------------------------------------------------------
// Launch
// ---------------------------------------------------------------------------
extern "C" void kernel_launch(void* const* d_in, const int* in_sizes, int n_in,
                              void* d_out, int out_size, void* d_ws, size_t ws_size,
                              hipStream_t stream) {
  (void)in_sizes; (void)n_in; (void)out_size; (void)ws_size;
  const float* x     = (const float*)d_in[0];
  const float* s_sty = (const float*)d_in[1];
  const float* wf    = (const float*)d_in[2];
  const float* bfv   = (const float*)d_in[3];
  const float* wg    = (const float*)d_in[4];
  const float* bg    = (const float*)d_in[5];
  const float* wh    = (const float*)d_in[6];
  const float* bh    = (const float*)d_in[7];
  const float* wk    = (const float*)d_in[8];
  const float* bk    = (const float*)d_in[9];
  float* out = (float*)d_out;

  char* w = (char*)d_ws;
  constexpr size_t SZ_BC_F = (size_t)BB * CC * sizeof(float);
  size_t off = 0;
  float* gsum   = (float*)(w + off); off += SZ_BC_F;
  float* gsumsq = (float*)(w + off); off += SZ_BC_F;
  float* muG    = (float*)(w + off); off += SZ_BC_F;
  float* rstdG  = (float*)(w + off); off += SZ_BC_F;
  float* muF    = (float*)(w + off); off += SZ_BC_F;
  float* rstdF  = (float*)(w + off); off += SZ_BC_F;
  float* bgeff  = (float*)(w + off); off += SZ_BC_F;
  float* wft    = (float*)(w + off); off += (size_t)CC * CC * sizeof(float);
  bf16* whb     = (bf16*)(w + off);  off += (size_t)CC * CC * sizeof(bf16);
  bf16* wkb     = (bf16*)(w + off);  off += (size_t)CC * CC * sizeof(bf16);
  bf16* wgeff   = (bf16*)(w + off);  off += (size_t)BB * CC * CC * sizeof(bf16);
  bf16* fpt     = (bf16*)(w + off);  off += (size_t)BB * LX * CC * sizeof(bf16);
  bf16* otb     = (bf16*)(w + off);  off += (size_t)BB * LX * CC * sizeof(bf16);

  hipMemsetAsync(gsum, 0, 2 * SZ_BC_F, stream);
  k_stats_g   <<<dim3(BB, LS / 256), 256, 0, stream>>>(s_sty, gsum, gsumsq);
  k_finalize_g<<<BB, 256, 0, stream>>>(gsum, gsumsq, muG, rstdG);
  k_stats_f   <<<BB, 256, 0, stream>>>(x, muF, rstdF);
  k_wconv     <<<CC, 256, 0, stream>>>(wf, wh, wk, wft, whb, wkb);
  k_prep_wg   <<<dim3(BB, CC), 256, 0, stream>>>(wg, bg, muG, rstdG, wgeff, bgeff);
  k_fpt       <<<BB * LX, 256, 0, stream>>>(x, wft, bfv, muF, rstdF, fpt);
  k_attn      <<<dim3(BB, LX / LT), 256, 0, stream>>>(s_sty, fpt, wgeff, bgeff, whb, bh, otb);
  k_final     <<<dim3(BB, 2), 256, 0, stream>>>(wkb, otb, bk, x, out);
}
